// SelfAttention_48473000902937
// MI455X (gfx1250) — matrix operand order repaired and checked by reading, not compiled, not run on hardware
//
#include <hip/hip_runtime.h>
#include <hip/hip_bf16.h>

// ---------------- problem constants ----------------
#define BB 4
#define SS 2048
#define EE 1024
#define HH 16
#define DD 64
#define MM (BB * SS)        // 8192 rows for all GEMMs

typedef __bf16        v16bf __attribute__((ext_vector_type(16)));
typedef float         v8f   __attribute__((ext_vector_type(8)));
typedef unsigned int  v4u   __attribute__((ext_vector_type(4)));
typedef unsigned int  v2u   __attribute__((ext_vector_type(2)));
typedef float         v4f   __attribute__((ext_vector_type(4)));

union Frag16 { v16bf bf; v4u q[2]; };

// fp32 -> bf16 round-to-nearest-even (bit pattern as ushort)
__device__ __forceinline__ unsigned short f2bf(float f) {
    unsigned int u = __float_as_uint(f);
    unsigned int r = u + 0x7FFFu + ((u >> 16) & 1u);
    return (unsigned short)(r >> 16);
}

// A fragment 16x32 bf16 (M x K), source row-major, row stride ld (elements).
// Lanes 0-15 -> rows, kb=0; lanes 16-31 -> same rows, kb=8.
// VGPR 0..3 hold K=kb..kb+7, VGPR 4..7 hold K=16+kb..16+kb+7 -> two 16B loads.
__device__ __forceinline__ v16bf load_frag_a(const unsigned short* base, int ld, int lane) {
    const int row = lane & 15;
    const int kb  = (lane & 16) ? 8 : 0;
    const unsigned short* p = base + row * ld + kb;
    Frag16 f;
    f.q[0] = *(const v4u*)(p);
    f.q[1] = *(const v4u*)(p + 16);
    return f.bf;
}

// B fragment 32x16 bf16 (K x N): lane n holds column n; lanes 0-15: K=0..7,16..23,
// lanes 16-31: K=8..15,24..31.  Column n contiguous along K at base + n*ld.
__device__ __forceinline__ v16bf load_frag_b(const unsigned short* base, int ld, int lane) {
    const int col = lane & 15;
    const int kb  = (lane & 16) ? 8 : 0;
    const unsigned short* p = base + col * ld + kb;
    Frag16 f;
    f.q[0] = *(const v4u*)(p);
    f.q[1] = *(const v4u*)(p + 16);
    return f.bf;
}

__device__ __forceinline__ v8f wmma_bf16(v16bf a, v16bf b, v8f c) {
    return __builtin_amdgcn_wmma_f32_16x16x32_bf16(false, a, false, b, (short)0, c, false, false);
}

// ---------------- fp32 -> bf16 conversion (4 elems / thread) ----------------
__global__ void __launch_bounds__(256)
cvt_bf16_kernel(const float* __restrict__ in, unsigned short* __restrict__ out, int n4) {
    int i = blockIdx.x * blockDim.x + threadIdx.x;
    if (i >= n4) return;
    v4f x = *(const v4f*)(in + (size_t)i * 4);
    v2u p;
    p[0] = (unsigned)f2bf(x[0]) | ((unsigned)f2bf(x[1]) << 16);
    p[1] = (unsigned)f2bf(x[2]) | ((unsigned)f2bf(x[3]) << 16);
    *(v2u*)(out + (size_t)i * 4) = p;
}

// ---------------- projection GEMM: Y = X @ W^T, bf16 in/out -----------------
// Wave tile: 32x64 (2 M-subtiles x 4 N-subtiles = 8 accumulators), software-
// pipelined fragment double-buffering so loads overlap WMMA issue.
// mode 0: out[((b*H+h)*S+s)*D + d]   ([B,H,S,D], used for Q and K)
// mode 2: out[((b*H+h)*D+d)*S + s]   ([B,H,D,S], transposed V)
__global__ void __launch_bounds__(256)
proj_kernel(const unsigned short* __restrict__ X, const unsigned short* __restrict__ W,
            unsigned short* __restrict__ out, int mode) {
    const int lane = threadIdx.x & 31;
    const int wid  = blockIdx.x * (blockDim.x >> 5) + (threadIdx.x >> 5);
    const int NT   = EE / 64;            // 16 N-strips
    const int tm   = wid / NT;           // 32-row M tile
    const int n0   = (wid % NT) * 64;    // N strip base
    const unsigned short* XrowL = X + (size_t)tm * 32 * EE;
    const unsigned short* XrowH = XrowL + (size_t)16 * EE;

    v8f cL0 = {}, cL1 = {}, cL2 = {}, cL3 = {};
    v8f cH0 = {}, cH1 = {}, cH2 = {}, cH3 = {};

    // prologue: fragments for k0 = 0
    v16bf aL = load_frag_a(XrowL, EE, lane);
    v16bf aH = load_frag_a(XrowH, EE, lane);
    v16bf b0 = load_frag_b(W + (size_t)(n0 +  0) * EE, EE, lane);
    v16bf b1 = load_frag_b(W + (size_t)(n0 + 16) * EE, EE, lane);
    v16bf b2 = load_frag_b(W + (size_t)(n0 + 32) * EE, EE, lane);
    v16bf b3 = load_frag_b(W + (size_t)(n0 + 48) * EE, EE, lane);

    for (int k0 = 0; k0 < EE; k0 += 32) {
        // issue next k-step's loads first (wraps on last iter; values unused)
        const int kn = (k0 + 32) & (EE - 1);
        v16bf aLn = load_frag_a(XrowL + kn, EE, lane);
        v16bf aHn = load_frag_a(XrowH + kn, EE, lane);
        v16bf b0n = load_frag_b(W + (size_t)(n0 +  0) * EE + kn, EE, lane);
        v16bf b1n = load_frag_b(W + (size_t)(n0 + 16) * EE + kn, EE, lane);
        v16bf b2n = load_frag_b(W + (size_t)(n0 + 32) * EE + kn, EE, lane);
        v16bf b3n = load_frag_b(W + (size_t)(n0 + 48) * EE + kn, EE, lane);

        cL0 = wmma_bf16(aL, b0, cL0);  cH0 = wmma_bf16(aH, b0, cH0);
        cL1 = wmma_bf16(aL, b1, cL1);  cH1 = wmma_bf16(aH, b1, cH1);
        cL2 = wmma_bf16(aL, b2, cL2);  cH2 = wmma_bf16(aH, b2, cH2);
        cL3 = wmma_bf16(aL, b3, cL3);  cH3 = wmma_bf16(aH, b3, cH3);

        aL = aLn; aH = aHn; b0 = b0n; b1 = b1n; b2 = b2n; b3 = b3n;
    }

    const int hi   = (lane & 16) ? 8 : 0;
    const int coll = lane & 15;
    v8f accs[8] = {cL0, cL1, cL2, cL3, cH0, cH1, cH2, cH3};
#pragma unroll
    for (int half = 0; half < 2; ++half) {
#pragma unroll
        for (int t = 0; t < 4; ++t) {
            int n = n0 + t * 16 + coll;
            int h = n >> 6, d = n & 63;
#pragma unroll
            for (int r = 0; r < 8; ++r) {
                int m = tm * 32 + half * 16 + r + hi;
                int b = m >> 11, s = m & (SS - 1);
                int idx;
                if (mode == 2) idx = ((b * HH + h) * DD + d) * SS + s;
                else           idx = ((b * HH + h) * SS + s) * DD + d;
                out[idx] = f2bf(accs[half * 4 + t][r]);
            }
        }
    }
}

// ---------------- flash-style attention ----------------
// Q,K: [B,H,S,D] bf16.  VT: [B,H,D,S] bf16.  ctx: [B,S,E] bf16.
// One wave per 16-query tile, 32 keys/iter with online softmax.
// Pipelining: V frags for the current chunk are issued before the logit WMMAs,
// next chunk's K frags right after them -> both fly during the softmax VALU.
__global__ void __launch_bounds__(256)
attn_kernel(const unsigned short* __restrict__ Q, const unsigned short* __restrict__ K,
            const unsigned short* __restrict__ VT, unsigned short* __restrict__ ctx) {
    __shared__ __align__(16) unsigned short plds[8][16 * 32];   // 1KB per wave
    const int lane = threadIdx.x & 31;
    const int w    = threadIdx.x >> 5;
    const int wid  = blockIdx.x * 8 + w;
    const int QT   = SS / 16;            // 128 query tiles per (b,h)
    const int bh   = wid / QT;
    const int qt   = wid % QT;
    const int b    = bh / HH, h = bh % HH;
    const unsigned short* Qb = Q  + ((size_t)bh * SS + qt * 16) * DD;
    const unsigned short* Kb = K  + (size_t)bh * SS * DD;
    const unsigned short* Vb = VT + (size_t)bh * DD * SS;
    unsigned short* myl = &plds[w][0];

    // Q fragments hoisted out of the key loop (K dim = D = 64 -> two k-steps)
    v16bf qa0 = load_frag_a(Qb,      DD, lane);
    v16bf qa1 = load_frag_a(Qb + 32, DD, lane);

    v8f o0 = {}, o1 = {}, o2 = {}, o3 = {};
    float mrow[8], lrow[8];
#pragma unroll
    for (int r = 0; r < 8; ++r) { mrow[r] = -1e30f; lrow[r] = 0.0f; }

    const int hi   = (lane & 16) ? 8 : 0;
    const int coll = lane & 15;

    // prologue: K fragments for chunk 0
    v16bf kf0 = load_frag_b(Kb + 0 * DD,       DD, lane);
    v16bf kf1 = load_frag_b(Kb + 0 * DD + 32,  DD, lane);
    v16bf kf2 = load_frag_b(Kb + 16 * DD,      DD, lane);
    v16bf kf3 = load_frag_b(Kb + 16 * DD + 32, DD, lane);

    for (int kc = 0; kc < SS; kc += 32) {
        // V fragments for the current chunk: issue early, consumed last
        v16bf vf0 = load_frag_b(Vb + (size_t)( 0) * SS + kc, SS, lane);
        v16bf vf1 = load_frag_b(Vb + (size_t)(16) * SS + kc, SS, lane);
        v16bf vf2 = load_frag_b(Vb + (size_t)(32) * SS + kc, SS, lane);
        v16bf vf3 = load_frag_b(Vb + (size_t)(48) * SS + kc, SS, lane);

        // logits for 32 keys: two 16x16 C tiles, K=64 via two WMMA k-steps
        v8f s0 = {}, s1 = {};
        s0 = wmma_bf16(qa0, kf0, s0);
        s0 = wmma_bf16(qa1, kf1, s0);
        s1 = wmma_bf16(qa0, kf2, s1);
        s1 = wmma_bf16(qa1, kf3, s1);

        // prefetch next chunk's K fragments (wraps on last iter; unused)
        const int kn = (kc + 32) & (SS - 1);
        kf0 = load_frag_b(Kb + (size_t)kn * DD,            DD, lane);
        kf1 = load_frag_b(Kb + (size_t)kn * DD + 32,       DD, lane);
        kf2 = load_frag_b(Kb + (size_t)(kn + 16) * DD,      DD, lane);
        kf3 = load_frag_b(Kb + (size_t)(kn + 16) * DD + 32, DD, lane);

        float osc[8];
#pragma unroll
        for (int r = 0; r < 8; ++r) {
            // row max across the 16-lane group holding this row's 16 columns
            float v = fmaxf(s0[r], s1[r]);
            v = fmaxf(v, __shfl_xor(v, 1, 32));
            v = fmaxf(v, __shfl_xor(v, 2, 32));
            v = fmaxf(v, __shfl_xor(v, 4, 32));
            v = fmaxf(v, __shfl_xor(v, 8, 32));
            float mn = fmaxf(mrow[r], v);
            float sc = __expf(mrow[r] - mn);
            float p0 = __expf(s0[r] - mn);
            float p1 = __expf(s1[r] - mn);
            float rs = p0 + p1;
            rs += __shfl_xor(rs, 1, 32);
            rs += __shfl_xor(rs, 2, 32);
            rs += __shfl_xor(rs, 4, 32);
            rs += __shfl_xor(rs, 8, 32);
            lrow[r] = lrow[r] * sc + rs;
            mrow[r] = mn;
            osc[r]  = sc;
            // stage P (C-layout) into LDS as a 16x32 bf16 tile
            int row = r + hi;
            myl[row * 32 + coll]      = f2bf(p0);
            myl[row * 32 + 16 + coll] = f2bf(p1);
        }
        // rescale running output
#pragma unroll
        for (int r = 0; r < 8; ++r) {
            o0[r] *= osc[r]; o1[r] *= osc[r]; o2[r] *= osc[r]; o3[r] *= osc[r];
        }
        // re-read P in A-layout (DS ops from the same wave are in-order)
        v16bf pa = load_frag_a(myl, 32, lane);
        // O += P @ V : VT columns (fixed d) are contiguous along keys
        o0 = wmma_bf16(pa, vf0, o0);
        o1 = wmma_bf16(pa, vf1, o1);
        o2 = wmma_bf16(pa, vf2, o2);
        o3 = wmma_bf16(pa, vf3, o3);
    }

    // finalize: softmax normalization then /sqrt(D)=8 (reference scales after softmax)
    v8f oo[4] = {o0, o1, o2, o3};
#pragma unroll
    for (int t = 0; t < 4; ++t) {
        int d = t * 16 + coll;
        int e = h * DD + d;
#pragma unroll
        for (int r = 0; r < 8; ++r) {
            int srow = qt * 16 + r + hi;
            float val = oo[t][r] / (lrow[r] * 8.0f);
            ctx[((size_t)b * SS + srow) * EE + e] = f2bf(val);
        }
    }
}

// ---------------- output GEMM: out = ctx @ Wo^T + bo, fp32 out --------------
// Same 32x64 pipelined wave tile as proj_kernel.
__global__ void __launch_bounds__(256)
gemm_out_kernel(const unsigned short* __restrict__ X, const unsigned short* __restrict__ W,
                const float* __restrict__ bo, float* __restrict__ out) {
    const int lane = threadIdx.x & 31;
    const int wid  = blockIdx.x * (blockDim.x >> 5) + (threadIdx.x >> 5);
    const int NT   = EE / 64;
    const int tm   = wid / NT;
    const int n0   = (wid % NT) * 64;
    const unsigned short* XrowL = X + (size_t)tm * 32 * EE;
    const unsigned short* XrowH = XrowL + (size_t)16 * EE;

    v8f cL0 = {}, cL1 = {}, cL2 = {}, cL3 = {};
    v8f cH0 = {}, cH1 = {}, cH2 = {}, cH3 = {};

    v16bf aL = load_frag_a(XrowL, EE, lane);
    v16bf aH = load_frag_a(XrowH, EE, lane);
    v16bf b0 = load_frag_b(W + (size_t)(n0 +  0) * EE, EE, lane);
    v16bf b1 = load_frag_b(W + (size_t)(n0 + 16) * EE, EE, lane);
    v16bf b2 = load_frag_b(W + (size_t)(n0 + 32) * EE, EE, lane);
    v16bf b3 = load_frag_b(W + (size_t)(n0 + 48) * EE, EE, lane);

    for (int k0 = 0; k0 < EE; k0 += 32) {
        const int kn = (k0 + 32) & (EE - 1);
        v16bf aLn = load_frag_a(XrowL + kn, EE, lane);
        v16bf aHn = load_frag_a(XrowH + kn, EE, lane);
        v16bf b0n = load_frag_b(W + (size_t)(n0 +  0) * EE + kn, EE, lane);
        v16bf b1n = load_frag_b(W + (size_t)(n0 + 16) * EE + kn, EE, lane);
        v16bf b2n = load_frag_b(W + (size_t)(n0 + 32) * EE + kn, EE, lane);
        v16bf b3n = load_frag_b(W + (size_t)(n0 + 48) * EE + kn, EE, lane);

        cL0 = wmma_bf16(aL, b0, cL0);  cH0 = wmma_bf16(aH, b0, cH0);
        cL1 = wmma_bf16(aL, b1, cL1);  cH1 = wmma_bf16(aH, b1, cH1);
        cL2 = wmma_bf16(aL, b2, cL2);  cH2 = wmma_bf16(aH, b2, cH2);
        cL3 = wmma_bf16(aL, b3, cL3);  cH3 = wmma_bf16(aH, b3, cH3);

        aL = aLn; aH = aHn; b0 = b0n; b1 = b1n; b2 = b2n; b3 = b3n;
    }

    const int hi   = (lane & 16) ? 8 : 0;
    const int coll = lane & 15;
    v8f accs[8] = {cL0, cL1, cL2, cL3, cH0, cH1, cH2, cH3};
#pragma unroll
    for (int half = 0; half < 2; ++half) {
#pragma unroll
        for (int t = 0; t < 4; ++t) {
            int n = n0 + t * 16 + coll;
            float bias = bo[n];
#pragma unroll
            for (int r = 0; r < 8; ++r) {
                int m = tm * 32 + half * 16 + r + hi;
                out[(size_t)m * EE + n] = accs[half * 4 + t][r] + bias;
            }
        }
    }
}

// ---------------- host-side launcher ----------------
extern "C" void kernel_launch(void* const* d_in, const int* in_sizes, int n_in,
                              void* d_out, int out_size, void* d_ws, size_t ws_size,
                              hipStream_t stream) {
    const float* values  = (const float*)d_in[0];
    const float* keys    = (const float*)d_in[1];
    const float* queries = (const float*)d_in[2];
    const float* Wv      = (const float*)d_in[3];
    const float* Wk      = (const float*)d_in[4];
    const float* Wq      = (const float*)d_in[5];
    const float* Wo      = (const float*)d_in[6];
    const float* bo      = (const float*)d_in[7];
    float* out = (float*)d_out;

    const size_t SZ_X = (size_t)MM * EE;            // 8388608 activation elems
    const size_t SZ_W = (size_t)EE * EE;            // 1048576 weight elems
    const size_t SZ_H = (size_t)BB * HH * SS * DD;  // per-head tensor elems

    unsigned short* ws  = (unsigned short*)d_ws;
    unsigned short* Xbf = ws;                         // values|keys|queries bf16
    unsigned short* Wbf = Xbf + 3 * SZ_X;             // Wv|Wk|Wq|Wo bf16
    unsigned short* Qh  = Wbf + 4 * SZ_W;             // [B,H,S,D]
    unsigned short* Kh  = Qh + SZ_H;                  // [B,H,S,D]
    unsigned short* VTh = Kh + SZ_H;                  // [B,H,D,S]
    unsigned short* Ctx = VTh + SZ_H;                 // [B,S,E]

    // 1) fp32 -> bf16 conversions
    {
        int gx = (int)(SZ_X / 4 / 256);
        cvt_bf16_kernel<<<gx, 256, 0, stream>>>(values,  Xbf + 0 * SZ_X, (int)(SZ_X / 4));
        cvt_bf16_kernel<<<gx, 256, 0, stream>>>(keys,    Xbf + 1 * SZ_X, (int)(SZ_X / 4));
        cvt_bf16_kernel<<<gx, 256, 0, stream>>>(queries, Xbf + 2 * SZ_X, (int)(SZ_X / 4));
        int gw = (int)(SZ_W / 4 / 256);
        cvt_bf16_kernel<<<gw, 256, 0, stream>>>(Wv, Wbf + 0 * SZ_W, (int)(SZ_W / 4));
        cvt_bf16_kernel<<<gw, 256, 0, stream>>>(Wk, Wbf + 1 * SZ_W, (int)(SZ_W / 4));
        cvt_bf16_kernel<<<gw, 256, 0, stream>>>(Wq, Wbf + 2 * SZ_W, (int)(SZ_W / 4));
        cvt_bf16_kernel<<<gw, 256, 0, stream>>>(Wo, Wbf + 3 * SZ_W, (int)(SZ_W / 4));
    }

    // 2) projections: waves = (M/32)*(E/64) = 4096 -> 512 blocks of 8 waves
    const int gemm_blocks = (MM / 32) * (EE / 64) / 8;
    proj_kernel<<<gemm_blocks, 256, 0, stream>>>(Xbf + 2 * SZ_X, Wbf + 2 * SZ_W, Qh, 0);
    proj_kernel<<<gemm_blocks, 256, 0, stream>>>(Xbf + 1 * SZ_X, Wbf + 1 * SZ_W, Kh, 0);
    proj_kernel<<<gemm_blocks, 256, 0, stream>>>(Xbf + 0 * SZ_X, Wbf + 0 * SZ_W, VTh, 2);

    // 3) attention: waves = B*H*(S/16) = 8192 -> 1024 blocks
    attn_kernel<<<(BB * HH * (SS / 16)) / 8, 256, 0, stream>>>(Qh, Kh, VTh, Ctx);

    // 4) output projection + bias (fp32 out)
    gemm_out_kernel<<<gemm_blocks, 256, 0, stream>>>(Ctx, Wbf + 3 * SZ_W, bo, out);
}
